// NNUE_39513699123399
// MI455X (gfx1250) — compile-verified, matrix operand order checked by reading, not run on hardware
//
#include <hip/hip_runtime.h>
#include <math.h>

typedef _Float16 v16h  __attribute__((ext_vector_type(16)));
typedef _Float16 half8 __attribute__((ext_vector_type(8)));
typedef float    v8f   __attribute__((ext_vector_type(8)));
typedef float    f32x8 __attribute__((ext_vector_type(8)));

#define N_IN    12288
#define H0      256
#define B_TOTAL 8192
#define ROWS    32          // batch rows per block
#define KC      128         // K-chunk (4 WMMA k-steps)
#define NCHUNK  (N_IN / KC) // 96
#define FSTR    136         // padded f16 stride for feature tiles (bank-friendly)
#define O0STR   520         // padded f16 stride for o0 (512 cols)
#define O1STR   40
#define O2STR   33

static __device__ __forceinline__ v8f wmma_f16(v16h a, v16h b, v8f c) {
  return __builtin_amdgcn_wmma_f32_16x16x32_f16(
      /*neg_a=*/false, a, /*neg_b=*/false, b,
      /*c_mod=*/(short)0, c, /*reuse_a=*/false, /*reuse_b=*/false);
}

// Fragment from f32 global memory: halves [0..7]=p[0..7], [8..15]=p[16..23]
static __device__ __forceinline__ v16h frag_from_f32(const float* __restrict__ p) {
  half8 lo = __builtin_convertvector(*(const f32x8*)p,        half8);
  half8 hi = __builtin_convertvector(*(const f32x8*)(p + 16), half8);
  return __builtin_shufflevector(lo, hi, 0,1,2,3,4,5,6,7,8,9,10,11,12,13,14,15);
}

// Fragment from f16 memory (LDS or global): two contiguous 16B runs (b128 x2)
static __device__ __forceinline__ v16h frag_from_f16(const _Float16* p) {
  half8 lo = *(const half8*)p;
  half8 hi = *(const half8*)(p + 16);
  return __builtin_shufflevector(lo, hi, 0,1,2,3,4,5,6,7,8,9,10,11,12,13,14,15);
}

// ---------------- ft_w f32 -> f16 one-shot converter ----------------
__global__ __launch_bounds__(256) void cvt_ftw_kernel(const float* __restrict__ src,
                                                      _Float16* __restrict__ dst) {
  const size_t i = ((size_t)blockIdx.x * 256 + threadIdx.x) * 8;
  f32x8 v = *(const f32x8*)(src + i);
  *(half8*)(dst + i) = __builtin_convertvector(v, half8);
}

// ---------------- fused NNUE kernel ----------------
template <bool W16>
__global__ __launch_bounds__(256) void nnue_fused_kernel(
    const float* __restrict__ wfeat, const float* __restrict__ bfeat,
    const unsigned char* __restrict__ side,
    const float* __restrict__ ft_w, const _Float16* __restrict__ ft_w16,
    const float* __restrict__ ft_b,
    const float* __restrict__ l1_w, const float* __restrict__ l1_b,
    const float* __restrict__ l2_w, const float* __restrict__ l2_b,
    const float* __restrict__ l3_w, const float* __restrict__ l3_b,
    float* __restrict__ out)
{
  __shared__ _Float16 sFeat[2][ROWS][FSTR];   // staged feature chunks (f16)
  __shared__ _Float16 sO0[ROWS][O0STR];       // concat+clipped FT output (f16)
  __shared__ _Float16 sO1[ROWS][O1STR];       // l1 output (f16)
  __shared__ float    sO2[ROWS][O2STR];       // l2 output (f32)
  __shared__ int      sSide[ROWS];

  const int tid    = threadIdx.x;
  const int wave   = tid >> 5;
  const int lane   = tid & 31;
  const int n16    = lane & 15;
  const int koff8  = (lane >> 4) * 8;         // K-half selector per WMMA layout
  const int rowbase = blockIdx.x * ROWS;

  if (tid < ROWS) sSide[tid] = (int)side[rowbase + tid];

  // FT accumulators: [color][mtile][ntile]  (color 0 = white)
  v8f acc[2][2][2] = {};

  // --- feature stager assignment: thread -> (color, row, 32-col strip) ---
  const int sc   = tid >> 7;        // color
  const int tt   = tid & 127;
  const int srow = tt >> 2;         // 0..31
  const int scol = (tt & 3) * 32;   // 0,32,64,96
  const float* sptr = (sc ? bfeat : wfeat) + (size_t)(rowbase + srow) * N_IN + scol;
  _Float16*    sdst = &sFeat[sc][srow][scol];

  // --- ft_w rows owned by this wave (2 N-tiles of 16 outputs) ---
  const float*    wrow0f = ft_w   + (size_t)(wave * 32 + n16) * N_IN;
  const float*    wrow1f = wrow0f + (size_t)16 * N_IN;
  const _Float16* wrow0h = ft_w16 + (size_t)(wave * 32 + n16) * N_IN;
  const _Float16* wrow1h = wrow0h + (size_t)16 * N_IN;

  for (int kc = 0; kc < NCHUNK; ++kc) {
    const int kbase = kc * KC;
    __syncthreads();
    // stage 32x128 f32 -> f16 per color (streaming, non-temporal);
    // prefetch next chunk
    __builtin_prefetch(sptr + kbase + KC, 0, 0);
#pragma unroll
    for (int j = 0; j < 4; ++j) {
      f32x8 v = __builtin_nontemporal_load((const f32x8*)(sptr + kbase + j * 8));
      *(half8*)(sdst + j * 8) = __builtin_convertvector(v, half8);
    }
    __syncthreads();

#pragma unroll
    for (int ks = 0; ks < 4; ++ks) {
      const int off = ks * 32 + koff8;
      v16h a00 = frag_from_f16(&sFeat[0][n16][off]);       // white, mtile 0
      v16h a01 = frag_from_f16(&sFeat[0][16 + n16][off]);  // white, mtile 1
      v16h a10 = frag_from_f16(&sFeat[1][n16][off]);       // black, mtile 0
      v16h a11 = frag_from_f16(&sFeat[1][16 + n16][off]);  // black, mtile 1
      const int kk = kbase + ks * 32 + koff8;
      v16h b0, b1;
      if constexpr (W16) {
        b0 = frag_from_f16(wrow0h + kk);
        b1 = frag_from_f16(wrow1h + kk);
      } else {
        b0 = frag_from_f32(wrow0f + kk);
        b1 = frag_from_f32(wrow1f + kk);
      }
      acc[0][0][0] = wmma_f16(a00, b0, acc[0][0][0]);
      acc[0][0][1] = wmma_f16(a00, b1, acc[0][0][1]);
      acc[0][1][0] = wmma_f16(a01, b0, acc[0][1][0]);
      acc[0][1][1] = wmma_f16(a01, b1, acc[0][1][1]);
      acc[1][0][0] = wmma_f16(a10, b0, acc[1][0][0]);
      acc[1][0][1] = wmma_f16(a10, b1, acc[1][0][1]);
      acc[1][1][0] = wmma_f16(a11, b0, acc[1][1][0]);
      acc[1][1][1] = wmma_f16(a11, b1, acc[1][1][1]);
    }
  }

  // --- bias + clip + side-dependent concat scatter into o0 ---
#pragma unroll
  for (int c = 0; c < 2; ++c) {
#pragma unroll
    for (int mt = 0; mt < 2; ++mt) {
#pragma unroll
      for (int nt = 0; nt < 2; ++nt) {
        const int h = wave * 32 + nt * 16 + n16;   // 0..255 within this color
        const float bias = ft_b[h];
#pragma unroll
        for (int r = 0; r < 8; ++r) {
          const int m = mt * 16 + r + (lane >> 4) * 8;   // C/D layout row
          float v = acc[c][mt][nt][r] + bias;
          v = fminf(fmaxf(v, 0.0f), 1.0f);
          // side==true  -> [white, black]; side==false -> [black, white]
          const int col = h + (((sSide[m] != 0) == (c == 0)) ? 0 : 256);
          sO0[m][col] = (_Float16)v;
        }
      }
    }
  }
  __syncthreads();

  // --- l1: (32 x 512) @ (512 -> 32), 4 output tiles on waves 0..3 ---
  const int mt = (wave >> 1) & 1;
  const int nt = wave & 1;
  if (wave < 4) {
    v8f t1 = {};
    const float* wr = l1_w + (size_t)(nt * 16 + n16) * (2 * H0);
#pragma unroll
    for (int kb = 0; kb < 16; ++kb) {
      v16h a = frag_from_f16(&sO0[mt * 16 + n16][kb * 32 + koff8]);
      v16h b = frag_from_f32(wr + kb * 32 + koff8);
      t1 = wmma_f16(a, b, t1);
    }
    const float bias = l1_b[nt * 16 + n16];
#pragma unroll
    for (int r = 0; r < 8; ++r) {
      const int m = mt * 16 + r + (lane >> 4) * 8;
      float v = fminf(fmaxf(t1[r] + bias, 0.0f), 1.0f);
      sO1[m][nt * 16 + n16] = (_Float16)v;
    }
  }
  __syncthreads();

  // --- l2: (32 x 32) @ (32 -> 32), single WMMA per tile ---
  if (wave < 4) {
    v8f t2 = {};
    v16h a = frag_from_f16(&sO1[mt * 16 + n16][koff8]);
    const float* wr = l2_w + (size_t)(nt * 16 + n16) * 32;
    v16h b = frag_from_f32(wr + koff8);
    t2 = wmma_f16(a, b, t2);
    const float bias = l2_b[nt * 16 + n16];
#pragma unroll
    for (int r = 0; r < 8; ++r) {
      const int m = mt * 16 + r + (lane >> 4) * 8;
      float v = fminf(fmaxf(t2[r] + bias, 0.0f), 1.0f);
      sO2[m][nt * 16 + n16] = v;
    }
  }
  __syncthreads();

  // --- l3 + sigmoid: one row per lane on wave 0 ---
  if (wave == 0) {
    float v = l3_b[0];
#pragma unroll
    for (int k = 0; k < 32; ++k) v += sO2[lane][k] * l3_w[k];
    // sigmoid((v * 300) / 200) = sigmoid(1.5 * v)
    out[rowbase + lane] = 1.0f / (1.0f + expf(-1.5f * v));
  }
}

extern "C" void kernel_launch(void* const* d_in, const int* in_sizes, int n_in,
                              void* d_out, int out_size, void* d_ws, size_t ws_size,
                              hipStream_t stream) {
  (void)in_sizes; (void)n_in; (void)out_size;
  const float*         wfeat = (const float*)d_in[0];
  const float*         bfeat = (const float*)d_in[1];
  const unsigned char* side  = (const unsigned char*)d_in[2];
  const float*         ft_w  = (const float*)d_in[3];
  const float*         ft_b  = (const float*)d_in[4];
  const float*         l1_w  = (const float*)d_in[5];
  const float*         l1_b  = (const float*)d_in[6];
  const float*         l2_w  = (const float*)d_in[7];
  const float*         l2_b  = (const float*)d_in[8];
  const float*         l3_w  = (const float*)d_in[9];
  const float*         l3_b  = (const float*)d_in[10];
  float* out = (float*)d_out;

  dim3 grid(B_TOTAL / ROWS);   // 256 blocks
  dim3 block(256);             // 8 waves (wave32)

  const size_t ftw_elems = (size_t)H0 * N_IN;          // 3,145,728
  const size_t need_ws   = ftw_elems * sizeof(_Float16); // 6.3 MB

  if (ws_size >= need_ws) {
    _Float16* ftw16 = (_Float16*)d_ws;
    // 8 elements per thread, 256 threads per block
    cvt_ftw_kernel<<<dim3((unsigned)(ftw_elems / (8 * 256))), dim3(256), 0, stream>>>(
        ft_w, ftw16);
    nnue_fused_kernel<true><<<grid, block, 0, stream>>>(
        wfeat, bfeat, side, ft_w, ftw16, ft_b, l1_w, l1_b, l2_w, l2_b, l3_w, l3_b, out);
  } else {
    nnue_fused_kernel<false><<<grid, block, 0, stream>>>(
        wfeat, bfeat, side, ft_w, (const _Float16*)nullptr, ft_b,
        l1_w, l1_b, l2_w, l2_b, l3_w, l3_b, out);
  }
}
